// PMoE_54056458388010
// MI455X (gfx1250) — compile-verified, matrix operand order checked by reading, not run on hardware
//
#include <hip/hip_runtime.h>
#include <hip/hip_bf16.h>
#include <math.h>

// ---------------------------------------------------------------------------
// PMoE forward for MI455X (gfx1250, wave32, WMMA).
//   E=8 experts, L=8 layers, R=4 rank, H=4096, V=32000, B=2, S=1024 -> N=2048
// Dominant cost: LM head GEMM [2048,4096]x[4096,32000] = 537 GFLOP.
//  - bf16 WMMA (16x16x32, 16384 FLOP/instr) with fp32 accumulate.
//  - Weights stream as fp32 and convert to bf16 at LDS-fill (v_cvt_pk_bf16_f32):
//    pre-converting would ADD ~786MB of HBM traffic per call, worse roofline.
//  - Hidden states are emitted as bf16 by the layers kernel, so the A tile is
//    pulled into LDS with GLOBAL_LOAD_ASYNC_TO_LDS_B128 (ASYNCcnt).
//  - 64x64 wave tiles: 16 WMMAs per 8 fragment loads per k-step.
// ---------------------------------------------------------------------------

#define NT   2048      // B*S tokens
#define HD   4096
#define NE   8
#define NL   8
#define NR   4
#define NV   32000

#define BM   128       // token tile
#define BN   256       // vocab tile
#define BK   32        // k slab (one WMMA K)

typedef __attribute__((ext_vector_type(16))) __bf16 v16bf;
typedef __attribute__((ext_vector_type(8)))  float  v8f;
typedef int v4i_g __attribute__((vector_size(16)));   // matches builtin param type

// ---- async global->LDS 16B copy (gfx1250), guarded so the build never breaks
#if defined(__has_builtin)
#  if __has_builtin(__builtin_amdgcn_global_load_async_to_lds_b128)
#    define ASYNC_LDS_BUILTIN __builtin_amdgcn_global_load_async_to_lds_b128
#    define HAVE_ASYNC_LDS 1
#  else
#    define HAVE_ASYNC_LDS 0
#  endif
#else
#  define HAVE_ASYNC_LDS 0
#endif

__device__ __forceinline__ void copy16_to_lds(const __bf16* g, __bf16* l) {
#if HAVE_ASYNC_LDS
  // param0: global v4i* (non-const), param1: LDS v4i*, then imm offset, imm cpol
  ASYNC_LDS_BUILTIN((__attribute__((address_space(1))) v4i_g*)(void*)g,
                    (__attribute__((address_space(3))) v4i_g*)(void*)l,
                    0, 0);
#else
  *(float4*)l = *(const float4*)g;   // fallback: 16B through VGPRs
#endif
}

__device__ __forceinline__ void wait_async_copies() {
#if HAVE_ASYNC_LDS
#  if __has_builtin(__builtin_amdgcn_s_wait_asynccnt)
  __builtin_amdgcn_s_wait_asynccnt(0);
#  else
  asm volatile("s_wait_asynccnt 0x0" ::: "memory");
#  endif
#endif
}

// ---------------------------------------------------------------------------
// Kernel 1: router logits + double softmax. One wave per token.
// ---------------------------------------------------------------------------
__global__ void pmoe_router(const float* __restrict__ h,
                            const float* __restrict__ rw,
                            const float* __restrict__ rb,
                            float* __restrict__ probs,     // [NT,NE] (2nd softmax)
                            float* __restrict__ rl_out) {  // [NT,NE] (1st softmax)
  const int wid  = threadIdx.x >> 5;
  const int lane = threadIdx.x & 31;
  const int token = blockIdx.x * 8 + wid;
  if (token >= NT) return;

  const float* hp = h + (size_t)token * HD;
  float acc[NE];
#pragma unroll
  for (int e = 0; e < NE; ++e) acc[e] = 0.f;

  for (int c = lane; c < HD; c += 32) {
    float hv = hp[c];
#pragma unroll
    for (int e = 0; e < NE; ++e) acc[e] += hv * rw[e * HD + c];
  }
#pragma unroll
  for (int e = 0; e < NE; ++e) {
#pragma unroll
    for (int off = 16; off > 0; off >>= 1)
      acc[e] += __shfl_xor(acc[e], off, 32);
    acc[e] += rb[e];
  }

  float m1 = acc[0];
#pragma unroll
  for (int e = 1; e < NE; ++e) m1 = fmaxf(m1, acc[e]);
  float l1[NE]; float s1 = 0.f;
#pragma unroll
  for (int e = 0; e < NE; ++e) { l1[e] = __expf(acc[e] - m1); s1 += l1[e]; }
  float inv1 = 1.f / s1;
#pragma unroll
  for (int e = 0; e < NE; ++e) l1[e] *= inv1;

  float m2 = l1[0];
#pragma unroll
  for (int e = 1; e < NE; ++e) m2 = fmaxf(m2, l1[e]);
  float p2[NE]; float s2 = 0.f;
#pragma unroll
  for (int e = 0; e < NE; ++e) { p2[e] = __expf(l1[e] - m2); s2 += p2[e]; }
  float inv2 = 1.f / s2;

  if (lane == 0) {
#pragma unroll
    for (int e = 0; e < NE; ++e) {
      rl_out[(size_t)token * NE + e] = l1[e];
      probs [(size_t)token * NE + e] = p2[e] * inv2;
    }
  }
}

// ---------------------------------------------------------------------------
// Kernel 2: 8 layers of expert LoRA mixture + deep LoRA. One wave per token,
// h lives in LDS. Emits bf16 hidden states for the WMMA LM head.
// ---------------------------------------------------------------------------
__global__ void pmoe_layers(const float* __restrict__ h_in,
                            const float* __restrict__ eB,   // [E][L][R][H]
                            const float* __restrict__ eA,   // [E][L][H][R]
                            const float* __restrict__ dB,   // [L][R][H]
                            const float* __restrict__ dA,   // [L][H][R]
                            const float* __restrict__ probs,// [NT,E]
                            __bf16* __restrict__ h_out) {   // [NT,H] bf16
  __shared__ float hsm[2][HD];            // 32 KB of the 320 KB WGP LDS
  const int wid  = threadIdx.x >> 5;      // 0..1
  const int lane = threadIdx.x & 31;
  const int token = blockIdx.x * 2 + wid;
  if (token >= NT) return;
  float* hs = hsm[wid];

  float p[NE]; float psum = 0.f;
#pragma unroll
  for (int e = 0; e < NE; ++e) { p[e] = probs[(size_t)token * NE + e]; psum += p[e]; }

  const float* hp = h_in + (size_t)token * HD;
  for (int c = lane; c < HD; c += 32) hs[c] = hp[c];

  const float4* A4  = (const float4*)eA;  // [E][L][H] of float4 (R=4)
  const float4* dA4 = (const float4*)dA;  // [L][H] of float4

  for (int l = 0; l < NL; ++l) {
    float zw[NE][NR];
#pragma unroll
    for (int e = 0; e < NE; ++e) {
#pragma unroll
      for (int r = 0; r < NR; ++r) {
        const float* bp = eB + ((((size_t)e * NL + l) * NR + r) * HD) + lane;
        float s = 0.f;
        for (int i = 0; i < HD / 32; ++i) s += hs[lane + 32 * i] * bp[32 * i];
#pragma unroll
        for (int off = 16; off > 0; off >>= 1) s += __shfl_xor(s, off, 32);
        zw[e][r] = s * p[e];
      }
    }
    for (int i = 0; i < HD / 32; ++i) {
      const int c = lane + 32 * i;
      float d = 0.f;
#pragma unroll
      for (int e = 0; e < NE; ++e) {
        float4 a = A4[((size_t)e * NL + l) * HD + c];
        d += zw[e][0] * a.x + zw[e][1] * a.y + zw[e][2] * a.z + zw[e][3] * a.w;
      }
      hs[c] = hs[c] * psum + d;
    }
    float low[NR];
#pragma unroll
    for (int r = 0; r < NR; ++r) {
      const float* bp = dB + (((size_t)l * NR + r) * HD) + lane;
      float s = 0.f;
      for (int i = 0; i < HD / 32; ++i) s += hs[lane + 32 * i] * bp[32 * i];
#pragma unroll
      for (int off = 16; off > 0; off >>= 1) s += __shfl_xor(s, off, 32);
      low[r] = s;
    }
    for (int i = 0; i < HD / 32; ++i) {
      const int c = lane + 32 * i;
      float4 a = dA4[(size_t)l * HD + c];
      hs[c] += low[0] * a.x + low[1] * a.y + low[2] * a.z + low[3] * a.w;
    }
  }

  __bf16* op = h_out + (size_t)token * HD;
  for (int c = lane; c < HD; c += 32) op[c] = (__bf16)hs[c];
}

// ---------------------------------------------------------------------------
// Kernel 3: LM head GEMM  out[N,V] = h[N,H] * W[V,H]^T  with bf16 WMMA.
// Block tile 128x256x32; 8 waves as 2(M)x4(N), wave tile 64x64 = 4x4 WMMA
// accumulators (16 v_wmma per k-step, 8 fragment loads). A (bf16 hidden) is
// async-copied into LDS; B (fp32 weights) is loaded, converted with
// v_cvt_pk_bf16_f32, and staged to LDS.
// ---------------------------------------------------------------------------
__global__ void __launch_bounds__(256)
pmoe_lmhead(const __bf16* __restrict__ h2,  // [NT,H] bf16 post-layer hidden
            const float*  __restrict__ w,   // [V,H] fp32
            float* __restrict__ out) {      // [NT,V]
  __shared__ __bf16 As[BM][48];   // 128x32 tile, rows padded to 96B (32B aligned)
  __shared__ __bf16 Bs[BN][48];   // 256x32 tile

  const int tid   = threadIdx.x;
  const int wid   = tid >> 5;
  const int lane  = tid & 31;
  const int wm    = wid & 1;           // 0..1 -> 64-row stripe
  const int wn    = wid >> 1;          // 0..3 -> 64-col stripe
  const int lrow  = lane & 15;
  const int khalf = lane >> 4;

  const int mBase = blockIdx.x * BM;   // token tile (16)
  const int nBase = blockIdx.y * BN;   // vocab tile (125)

  const float4* w4 = (const float4*)w;

  v8f acc[4][4];
  const v8f vzero = {0.f, 0.f, 0.f, 0.f, 0.f, 0.f, 0.f, 0.f};
#pragma unroll
  for (int mt = 0; mt < 4; ++mt)
#pragma unroll
    for (int nt = 0; nt < 4; ++nt) acc[mt][nt] = vzero;

  for (int k0 = 0; k0 < HD; k0 += BK) {
    // ---- A: 128x32 bf16 = 512 x 16B chunks; 2 async copies per thread
#pragma unroll
    for (int j = 0; j < 2; ++j) {
      const int ch  = tid + 256 * j;
      const int row = ch >> 2;          // 0..127
      const int c8  = (ch & 3) * 8;     // 0,8,16,24 (bf16 units)
      copy16_to_lds(&h2[(size_t)(mBase + row) * HD + k0 + c8], &As[row][c8]);
    }
    // ---- B: 256x32 fp32 -> bf16; 8 float4 per thread
#pragma unroll
    for (int j = 0; j < 8; ++j) {
      const int idx = tid + 256 * j;
      const int row = idx >> 3;         // 0..255
      const int c4  = idx & 7;          // float4 column
      float4 bv = w4[(size_t)(nBase + row) * (HD / 4) + (k0 >> 2) + c4];
      __bf16* bp = &Bs[row][c4 * 4];
      bp[0] = (__bf16)bv.x; bp[1] = (__bf16)bv.y;
      bp[2] = (__bf16)bv.z; bp[3] = (__bf16)bv.w;
    }
    wait_async_copies();                // ASYNCcnt -> 0 before publishing tile
    __syncthreads();

    // ---- prefetch next weight slab into cache (global_prefetch_b8)
    if (k0 + BK < HD) {
      __builtin_prefetch(&w4[(size_t)(nBase + (tid >> 3)) * (HD / 4) + ((k0 + BK) >> 2) + (tid & 7)], 0, 1);
    }

    // ---- fragments: lane holds K = khalf*16 .. +15 of row (lane&15)
    v16bf afrag[4], bfrag[4];
#pragma unroll
    for (int mt = 0; mt < 4; ++mt)
      afrag[mt] = *(const v16bf*)&As[wm * 64 + mt * 16 + lrow][khalf * 16];
#pragma unroll
    for (int nt = 0; nt < 4; ++nt)
      bfrag[nt] = *(const v16bf*)&Bs[wn * 64 + nt * 16 + lrow][khalf * 16];

#pragma unroll
    for (int mt = 0; mt < 4; ++mt)
#pragma unroll
      for (int nt = 0; nt < 4; ++nt)
        acc[mt][nt] = __builtin_amdgcn_wmma_f32_16x16x32_bf16(
            false, afrag[mt], false, bfrag[nt],
            (short)0, acc[mt][nt], false, false);

    __syncthreads();
  }

  // ---- epilogue: C layout = VGPR r -> M = r + 8*(lane>=16), N = lane&15
#pragma unroll
  for (int mt = 0; mt < 4; ++mt) {
#pragma unroll
    for (int nt = 0; nt < 4; ++nt) {
      const int col = nBase + wn * 64 + nt * 16 + (lane & 15);
#pragma unroll
      for (int r = 0; r < 8; ++r) {
        const int row = mBase + wm * 64 + mt * 16 + r + 8 * khalf;
        out[(size_t)row * NV + col] = acc[mt][nt][r];
      }
    }
  }
}

// ---------------------------------------------------------------------------
extern "C" void kernel_launch(void* const* d_in, const int* in_sizes, int n_in,
                              void* d_out, int out_size, void* d_ws, size_t ws_size,
                              hipStream_t stream) {
  const float* h   = (const float*)d_in[0];  // [B,S,H]
  const float* rw  = (const float*)d_in[1];  // [E,H]
  const float* rb  = (const float*)d_in[2];  // [E]
  const float* eB  = (const float*)d_in[3];  // [E,L,R,H]
  const float* eA  = (const float*)d_in[4];  // [E,L,H,R]
  const float* dB  = (const float*)d_in[5];  // [L,R,H]
  const float* dA  = (const float*)d_in[6];  // [L,H,R]
  const float* lmw = (const float*)d_in[7];  // [V,H]

  float* out    = (float*)d_out;                         // lm_logits [NT,V]
  float* rl_out = out + (size_t)NT * NV;                 // router_logits [NT,E]

  // workspace: probs [NT,E] fp32 (64KB) then bf16 hidden [NT,H] (16.8MB)
  float*  probs = (float*)d_ws;
  __bf16* h2    = (__bf16*)(probs + (size_t)NT * NE);

  (void)in_sizes; (void)n_in; (void)out_size; (void)ws_size;

  pmoe_router<<<dim3(NT / 8), dim3(256), 0, stream>>>(h, rw, rb, probs, rl_out);
  pmoe_layers<<<dim3(NT / 2), dim3(64), 0, stream>>>(h, eB, eA, dB, dA, probs, h2);
  // vocab-major grid: the 16 token-tiles of a vocab tile are adjacent in
  // launch order, so weight tiles are reused through L2.
  pmoe_lmhead<<<dim3(NT / BM, NV / BN), dim3(256), 0, stream>>>(h2, lmw, out);
}
